// SplineCoupling_34789235097929
// MI455X (gfx1250) — matrix operand order repaired and checked by reading, not compiled, not run on hardware
//
#include <hip/hip_runtime.h>

// ---------------------------------------------------------------------------
// Spline coupling flow, fused for gfx1250 (MI455X).
//   Pass 1: pack W0/W1/Wp (f32) -> bf16 fragments in contiguous-k WMMA-B fragment layout (d_ws)
//   Pass 2: fused per-16-row-tile kernel:
//       GEMM0 (32->128) + LN + relu            (v_wmma_f32_16x16x32_bf16)
//       GEMM1 (128->128) + LN + relu + residual
//       GEMM2 (128->800) + bias -> LDS
//       softmax/softplus + monotone RQS spline, y + log_det out
//   Cross-lane reductions use ds_swizzle_b32 (single-instruction lane xor).
// ---------------------------------------------------------------------------

typedef __bf16 bf16_t;
typedef bf16_t v16bf __attribute__((ext_vector_type(16)));
typedef float  v8f   __attribute__((ext_vector_type(8)));

#define B_ROWS   262144
#define DFEAT    64
#define D2       32
#define HID      128
#define NOUT     800          // 32 * 25
#define NBINS    8

// packed bf16 weight layout in d_ws (ushort units)
#define OFF_W0   0            // 8 tiles * 512
#define OFF_W1   4096         // 8 tiles * 4 chunks * 512
#define OFF_WP   20480        // 50 tiles * 4 chunks * 512
#define PK_TOTAL 122880

__device__ __forceinline__ unsigned short f2bfb(float f) {
    unsigned u = __builtin_bit_cast(unsigned, f);
    u += 0x7FFFu + ((u >> 16) & 1u);          // round-to-nearest-even
    return (unsigned short)(u >> 16);
}
__device__ __forceinline__ unsigned packbf(float a, float b) {
    return (unsigned)f2bfb(a) | ((unsigned)f2bfb(b) << 16);
}

// lane-xor exchange via ds_swizzle_b32 (group-of-32 mode: and=0x1f, or=0, xor=m)
#if __has_builtin(__builtin_amdgcn_ds_swizzle)
template <int PATT>
__device__ __forceinline__ float swzf(float v) {
    return __builtin_bit_cast(float,
        __builtin_amdgcn_ds_swizzle(__builtin_bit_cast(int, v), PATT));
}
#else
template <int PATT>
__device__ __forceinline__ float swzf(float v) {
    return __shfl_xor(v, (PATT >> 10) & 0x1f, 32);
}
#endif

// sum across each 16-lane half (xor 1,2,4,8)
__device__ __forceinline__ float red16(float v) {
    v += swzf<0x041F>(v);
    v += swzf<0x081F>(v);
    v += swzf<0x101F>(v);
    v += swzf<0x201F>(v);
    return v;
}
// sum across all 32 lanes
__device__ __forceinline__ float red32(float v) {
    v = red16(v);
    v += swzf<0x401F>(v);
    return v;
}

union FragU { v16bf v; unsigned u[8]; uint4 q[2]; };

// ---------------------------------------------------------------------------
// Weight pack kernel: element e of lane L of fragment (tile t, k-chunk c) is
// B[k][n] with k = c*32 + (L>=16 ? 16 : 0) + e, n = t*16 + (L&15)
// (contiguous k order, same as the A fragments). Stored e-fastest => lane reads 32B contig.
// ---------------------------------------------------------------------------
__global__ void pack_weights_k(const float* __restrict__ W0,
                               const float* __restrict__ W1,
                               const float* __restrict__ Wp,
                               unsigned short* __restrict__ pk) {
    int idx = blockIdx.x * blockDim.x + threadIdx.x;
    if (idx >= PK_TOTAL) return;
    float v;
    if (idx < OFF_W1) {                       // W0: 32 x 128
        int t = idx >> 9, r = idx & 511;
        int L = r >> 4, e = r & 15;
        int k = ((L >= 16) ? 16 : 0) + e;
        int n = t * 16 + (L & 15);
        v = W0[k * HID + n];
    } else if (idx < OFF_WP) {                // W1: 128 x 128
        int j = idx - OFF_W1;
        int f = j >> 9, r = j & 511;
        int t = f >> 2, c = f & 3;
        int L = r >> 4, e = r & 15;
        int k = c * 32 + ((L >= 16) ? 16 : 0) + e;
        int n = t * 16 + (L & 15);
        v = W1[k * HID + n];
    } else {                                  // Wp: 128 x 800
        int j = idx - OFF_WP;
        int f = j >> 9, r = j & 511;
        int t = f >> 2, c = f & 3;
        int L = r >> 4, e = r & 15;
        int k = c * 32 + ((L >= 16) ? 16 : 0) + e;
        int n = t * 16 + (L & 15);
        v = Wp[k * NOUT + n];
    }
    pk[idx] = f2bfb(v);
}

__device__ __forceinline__ float softplusf(float x) {
    return (x > 0.f) ? (x + __logf(1.f + __expf(-x)))
                     : __logf(1.f + __expf(x));
}

// ---------------------------------------------------------------------------
// Fused main kernel: one 16-row tile per 256-thread block (8 waves).
// ---------------------------------------------------------------------------
__global__ __launch_bounds__(256)
void spline_fused_k(const float* __restrict__ x,
                    const float* __restrict__ b0v, const float* __restrict__ g0v,
                    const float* __restrict__ be0v,
                    const float* __restrict__ b1v, const float* __restrict__ g1v,
                    const float* __restrict__ be1v,
                    const float* __restrict__ bpv,
                    const unsigned short* __restrict__ pk,
                    float* __restrict__ yout, float* __restrict__ ldout) {
    __shared__ float    xt[16 * 65];     // x tile, row stride 65 (bank-safe)
    __shared__ unsigned hb[16 * 66];     // h as bf16 pairs, row stride 66 dwords
    __shared__ float    raw[16 * 804];   // spline params, row stride 804
    __shared__ float    red0[8][16];
    __shared__ float    red1[8][16];
    __shared__ float2   stats[16];

    const int tid  = threadIdx.x;
    const int wave = tid >> 5;
    const int lane = tid & 31;
    const int l15  = lane & 15;
    const int hi   = lane >> 4;       // lane half
    const int off0 = hi * 8;          // accumulator row offset for this half
    const int row0 = blockIdx.x * 16;

    // ---- stage x tile (coalesced b128) ----
    {
        const float4* xg = reinterpret_cast<const float4*>(x + (size_t)row0 * DFEAT);
        float4 v = xg[tid];
        int g = tid * 4;
        int r = g >> 6, c = g & 63;
        xt[r * 65 + c + 0] = v.x;  xt[r * 65 + c + 1] = v.y;
        xt[r * 65 + c + 2] = v.z;  xt[r * 65 + c + 3] = v.w;
    }
    __syncthreads();

    const uint4* pk4 = reinterpret_cast<const uint4*>(pk);
    const int n0 = wave * 16 + l15;   // this wave's column (0..127)

    // =========================== GEMM0: xm @ W0 ===========================
    FragU a0;
    #pragma unroll
    for (int w = 0; w < 8; ++w) {
        int kb = hi * 16 + 2 * w;
        // xm[:,k] = x[:, 2k]
        a0.u[w] = packbf(xt[l15 * 65 + 2 * kb], xt[l15 * 65 + 2 * (kb + 1)]);
    }
    FragU bw;
    bw.q[0] = pk4[(OFF_W0 >> 3) + wave * 64 + lane * 2 + 0];
    bw.q[1] = pk4[(OFF_W0 >> 3) + wave * 64 + lane * 2 + 1];
    v8f acc;
    #pragma unroll
    for (int i = 0; i < 8; ++i) acc[i] = 0.f;
    acc = __builtin_amdgcn_wmma_f32_16x16x32_bf16(false, a0.v, false, bw.v,
                                                  (short)0, acc, false, false);
    {
        float bb = b0v[n0];
        #pragma unroll
        for (int i = 0; i < 8; ++i) acc[i] += bb;
    }
    // ---- LayerNorm 0 reduction (rows striped r+8*hi across v8f) ----
    #pragma unroll
    for (int r = 0; r < 8; ++r) {
        float s = red16(acc[r]);
        float q = red16(acc[r] * acc[r]);
        if (l15 == 0) { red0[wave][r + off0] = s; red1[wave][r + off0] = q; }
    }
    __syncthreads();
    if (tid < 16) {
        float S = 0.f, Q = 0.f;
        #pragma unroll
        for (int w = 0; w < 8; ++w) { S += red0[w][tid]; Q += red1[w][tid]; }
        float mu = S * (1.f / 128.f);
        float var = Q * (1.f / 128.f) - mu * mu;
        stats[tid] = make_float2(mu, rsqrtf(var + 1e-6f));
    }
    __syncthreads();
    float h1v[8];
    {
        float gg = g0v[n0], bb = be0v[n0];
        unsigned short* hbh = reinterpret_cast<unsigned short*>(hb);
        #pragma unroll
        for (int r = 0; r < 8; ++r) {
            int M = r + off0;
            float v = (acc[r] - stats[M].x) * stats[M].y * gg + bb;
            v = fmaxf(v, 0.f);
            h1v[r] = v;
            hbh[M * 132 + n0] = f2bfb(v);
        }
    }
    __syncthreads();

    // =========================== GEMM1: h1 @ W1 ===========================
    FragU A1[4];
    #pragma unroll
    for (int c = 0; c < 4; ++c)
        #pragma unroll
        for (int w = 0; w < 8; ++w) {
            int kb = hi * 16 + 2 * w;
            A1[c].u[w] = hb[l15 * 66 + ((c * 32 + kb) >> 1)];
        }
    v8f acc2;
    #pragma unroll
    for (int i = 0; i < 8; ++i) acc2[i] = 0.f;
    #pragma unroll
    for (int c = 0; c < 4; ++c) {
        FragU bf;
        int b4 = (OFF_W1 >> 3) + (wave * 4 + c) * 64 + lane * 2;
        bf.q[0] = pk4[b4 + 0];
        bf.q[1] = pk4[b4 + 1];
        acc2 = __builtin_amdgcn_wmma_f32_16x16x32_bf16(false, A1[c].v, false, bf.v,
                                                       (short)0, acc2, false, false);
    }
    {
        float bb = b1v[n0];
        #pragma unroll
        for (int i = 0; i < 8; ++i) acc2[i] += bb;
    }
    // ---- LayerNorm 1 reduction ----
    #pragma unroll
    for (int r = 0; r < 8; ++r) {
        float s = red16(acc2[r]);
        float q = red16(acc2[r] * acc2[r]);
        if (l15 == 0) { red0[wave][r + off0] = s; red1[wave][r + off0] = q; }
    }
    __syncthreads();
    if (tid < 16) {
        float S = 0.f, Q = 0.f;
        #pragma unroll
        for (int w = 0; w < 8; ++w) { S += red0[w][tid]; Q += red1[w][tid]; }
        float mu = S * (1.f / 128.f);
        float var = Q * (1.f / 128.f) - mu * mu;
        stats[tid] = make_float2(mu, rsqrtf(var + 1e-6f));
    }
    __syncthreads();
    {
        float gg = g1v[n0], bb = be1v[n0];
        unsigned short* hbh = reinterpret_cast<unsigned short*>(hb);
        #pragma unroll
        for (int r = 0; r < 8; ++r) {
            int M = r + off0;
            float v = (acc2[r] - stats[M].x) * stats[M].y * gg + bb;
            v = fmaxf(v, 0.f) + h1v[r];        // residual
            hbh[M * 132 + n0] = f2bfb(v);
        }
    }
    __syncthreads();

    // =========================== GEMM2: h2 @ Wp ===========================
    FragU A2[4];
    #pragma unroll
    for (int c = 0; c < 4; ++c)
        #pragma unroll
        for (int w = 0; w < 8; ++w) {
            int kb = hi * 16 + 2 * w;
            A2[c].u[w] = hb[l15 * 66 + ((c * 32 + kb) >> 1)];
        }
    for (int t2 = wave; t2 < 50; t2 += 8) {
        // prefetch next tile's packed-B fragments into cache (global_prefetch_b8)
        if (t2 + 8 < 50)
            __builtin_prefetch(pk + OFF_WP + (size_t)(t2 + 8) * 4 * 512 + lane * 64,
                               0, 1);
        v8f a3;
        #pragma unroll
        for (int i = 0; i < 8; ++i) a3[i] = 0.f;
        #pragma unroll
        for (int c = 0; c < 4; ++c) {
            FragU bf;
            int b4 = (OFF_WP >> 3) + (t2 * 4 + c) * 64 + lane * 2;
            bf.q[0] = pk4[b4 + 0];
            bf.q[1] = pk4[b4 + 1];
            a3 = __builtin_amdgcn_wmma_f32_16x16x32_bf16(false, A2[c].v, false, bf.v,
                                                         (short)0, a3, false, false);
        }
        int n2 = t2 * 16 + l15;
        float bb = bpv[n2];
        #pragma unroll
        for (int r = 0; r < 8; ++r)
            raw[(r + off0) * 804 + n2] = a3[r] + bb;
    }
    __syncthreads();

    // ====================== spline phase: 2 x 256 pairs ====================
    #pragma unroll
    for (int pass = 0; pass < 2; ++pass) {
        int p  = tid + pass * 256;
        int rr = p >> 5;         // row in tile
        int u  = p & 31;         // unmasked dim
        int base = rr * 804 + u * 25;

        float wr[8], hr[8], dr[9];
        #pragma unroll
        for (int i = 0; i < 8; ++i) { wr[i] = raw[base + i]; hr[i] = raw[base + 8 + i]; }
        #pragma unroll
        for (int i = 0; i < 9; ++i) dr[i] = raw[base + 16 + i];

        // softmax(widths), softmax(heights)
        float mw = wr[0], mh = hr[0];
        #pragma unroll
        for (int i = 1; i < 8; ++i) { mw = fmaxf(mw, wr[i]); mh = fmaxf(mh, hr[i]); }
        float we[8], he[8], sw = 0.f, sh = 0.f;
        #pragma unroll
        for (int i = 0; i < 8; ++i) {
            we[i] = __expf(wr[i] - mw); sw += we[i];
            he[i] = __expf(hr[i] - mh); sh += he[i];
        }
        float iw = 1.f / sw, ih = 1.f / sh;
        float wd[8], hd[8], cw[9], ch[9], dv[9];
        cw[0] = -3.f; ch[0] = -3.f;
        #pragma unroll
        for (int i = 0; i < 8; ++i) {
            wd[i] = (0.001f + 0.992f * we[i] * iw) * 6.0f;
            hd[i] = (0.001f + 0.992f * he[i] * ih) * 6.0f;
            cw[i + 1] = cw[i] + wd[i];
            ch[i + 1] = ch[i] + hd[i];
        }
        #pragma unroll
        for (int i = 0; i < 9; ++i) dv[i] = softplusf(dr[i]) + 0.001f;

        float xu = xt[rr * 65 + 2 * u + 1];
        bool inside = (xu > -3.f) && (xu < 3.f);
        float xc = fminf(fmaxf(xu, -3.f), 3.f);
        int cnt = 0;
        #pragma unroll
        for (int i = 0; i < 9; ++i) cnt += (xc >= cw[i]) ? 1 : 0;
        int idx = cnt - 1;
        idx = idx < 0 ? 0 : (idx > 7 ? 7 : idx);

        float w_k = wd[0], x_k = cw[0], h_k = hd[0], y_k = ch[0];
        float d_k = dv[0], d_k1 = dv[1];
        #pragma unroll
        for (int i = 1; i < 8; ++i) {
            bool sel = (idx == i);
            w_k = sel ? wd[i] : w_k;  x_k = sel ? cw[i]     : x_k;
            h_k = sel ? hd[i] : h_k;  y_k = sel ? ch[i]     : y_k;
            d_k = sel ? dv[i] : d_k;  d_k1 = sel ? dv[i + 1] : d_k1;
        }

        float s   = h_k / w_k;
        float th  = (xc - x_k) / w_k;
        float t1m = th * (1.f - th);
        float den = s + (d_k1 + d_k - 2.f * s) * t1m;
        float yin = y_k + h_k * (s * th * th + d_k * t1m) / den;
        float num = d_k1 * th * th + 2.f * s * t1m + d_k * (1.f - th) * (1.f - th);
        float ldi = 2.f * __logf(s) + __logf(num) - 2.f * __logf(den);

        float yv  = inside ? yin : xu;
        float ldv = inside ? ldi : 0.f;

        // write interleaved output pair (xm, yu)
        float2 o;
        o.x = xt[rr * 65 + 2 * u];
        o.y = yv;
        reinterpret_cast<float2*>(yout + (size_t)(row0 + rr) * DFEAT)[u] = o;

        // wave covers one row: reduce log_det over its 32 lanes
        ldv = red32(ldv);
        if (lane == 0) ldout[row0 + rr] = ldv;
    }
}

extern "C" void kernel_launch(void* const* d_in, const int* in_sizes, int n_in,
                              void* d_out, int out_size, void* d_ws, size_t ws_size,
                              hipStream_t stream) {
    (void)in_sizes; (void)n_in; (void)out_size; (void)ws_size;
    const float* x   = (const float*)d_in[0];
    const float* W0  = (const float*)d_in[1];
    const float* b0  = (const float*)d_in[2];
    const float* g0  = (const float*)d_in[3];
    const float* be0 = (const float*)d_in[4];
    const float* W1  = (const float*)d_in[5];
    const float* b1  = (const float*)d_in[6];
    const float* g1  = (const float*)d_in[7];
    const float* be1 = (const float*)d_in[8];
    const float* Wp  = (const float*)d_in[9];
    const float* bp  = (const float*)d_in[10];

    unsigned short* pk = (unsigned short*)d_ws;
    float* yout  = (float*)d_out;
    float* ldout = yout + (size_t)B_ROWS * DFEAT;

    pack_weights_k<<<(PK_TOTAL + 255) / 256, 256, 0, stream>>>(W0, W1, Wp, pk);
    spline_fused_k<<<B_ROWS / 16, 256, 0, stream>>>(
        x, b0, g0, be0, b1, g1, be1, bp, pk, yout, ldout);
}
